// V2TransformerLayer_63797444215497
// MI455X (gfx1250) — matrix operand order repaired and checked by reading, not compiled, not run on hardware
//
#include <hip/hip_runtime.h>
#include <hip/hip_bf16.h>
#include <math.h>

typedef __attribute__((ext_vector_type(16))) __bf16 v16bf;
typedef __attribute__((ext_vector_type(8)))  float  v8f;

#define D_MODEL 1024
#define N_HEADS 16
#define D_HEAD  64
#define D_FF    4096
#define BATCH   4
#define SEQ     2048
#define BT      (BATCH*SEQ)

union Frag { v16bf v; uint4 u[2]; };

// ---- CDNA5 async global->LDS copy helpers (ASYNCcnt-tracked, no VGPR data path)
__device__ __forceinline__ void async_g2l_b128(unsigned lds_addr, unsigned long long gaddr) {
  asm volatile("global_load_async_to_lds_b128 %0, %1, off" :: "v"(lds_addr), "v"(gaddr) : "memory");
}
__device__ __forceinline__ void async_g2l_b128_o16(unsigned lds_addr, unsigned long long gaddr) {
  // INST_OFFSET is added to BOTH the LDS and global addresses (ISA 08 §4.4)
  asm volatile("global_load_async_to_lds_b128 %0, %1, off offset:16" :: "v"(lds_addr), "v"(gaddr) : "memory");
}
__device__ __forceinline__ void wait_async4() {
#if __has_builtin(__builtin_amdgcn_s_wait_asynccnt)
  __builtin_amdgcn_s_wait_asynccnt(4);
#else
  asm volatile("s_wait_asynccnt 0x4" ::: "memory");
#endif
}
__device__ __forceinline__ void wait_async0() {
#if __has_builtin(__builtin_amdgcn_s_wait_asynccnt)
  __builtin_amdgcn_s_wait_asynccnt(0);
#else
  asm volatile("s_wait_asynccnt 0x0" ::: "memory");
#endif
}
__device__ __forceinline__ unsigned lds_off(const void* p) {
  // flat LDS-aperture address: low 32 bits are the DS byte address (ISA 00 §10.2)
  return (unsigned)(unsigned long long)p;
}

// ---------------- weight transpose + fp32->bf16 convert: Wt[n][k] = W[k][n]
__global__ void k_transpose_cvt(const float* __restrict__ W, __bf16* __restrict__ Wt,
                                int K, int N) {
  size_t i = (size_t)blockIdx.x * blockDim.x + threadIdx.x;
  if (i >= (size_t)K * N) return;
  int k = (int)(i / N), n = (int)(i % N);
  Wt[(size_t)n * K + k] = (__bf16)W[i];
}

// ---------------- row LayerNorm fp32 -> bf16 (one block per row of 1024)
__global__ __launch_bounds__(256) void k_layernorm(const float* __restrict__ x,
                                                   const float* __restrict__ g,
                                                   const float* __restrict__ b,
                                                   __bf16* __restrict__ out) {
  __shared__ float s1[256], s2[256];
  int row = blockIdx.x, tid = threadIdx.x;
  const float* xr = x + (size_t)row * D_MODEL;
  float v[4], sum = 0.f, sq = 0.f;
  for (int j = 0; j < 4; j++) { v[j] = xr[tid + j*256]; sum += v[j]; sq += v[j]*v[j]; }
  s1[tid] = sum; s2[tid] = sq; __syncthreads();
  for (int st = 128; st > 0; st >>= 1) {
    if (tid < st) { s1[tid] += s1[tid+st]; s2[tid] += s2[tid+st]; }
    __syncthreads();
  }
  float mu  = s1[0] * (1.f / D_MODEL);
  float var = s2[0] * (1.f / D_MODEL) - mu*mu;
  float rs  = rsqrtf(var + 1e-5f);
  for (int j = 0; j < 4; j++) {
    int c = tid + j*256;
    out[(size_t)row*D_MODEL + c] = (__bf16)((v[j]-mu)*rs*g[c] + b[c]);
  }
}

// ---------------- generic bf16 WMMA GEMM: C = A(MxK) @ Bt(NxK)^T + bias, epilogues
// BM=BN=128, BK=32; 256 threads = 8 waves (2x4); wave tile 64x32 = 4x2 WMMA frags.
// Double-buffered LDS, tiles staged with GLOBAL_LOAD_ASYNC_TO_LDS_B128.
constexpr int EPI_QK   = 0; // scale, store bf16 remapped [b,h,t,d]
constexpr int EPI_VT   = 1; // store bf16 remapped transposed [b,h,d,t]
constexpr int EPI_RES  = 2; // fp32 out = acc + bias + res
constexpr int EPI_GELU = 3; // bf16 out = gelu(acc + bias)

template <int EPI>
__global__ __launch_bounds__(256) void k_gemm(const __bf16* __restrict__ A,
                                              const __bf16* __restrict__ Bt,
                                              const float* __restrict__ bias,
                                              const float* __restrict__ res,
                                              float* __restrict__ outf,
                                              __bf16* __restrict__ outb,
                                              int M, int N, int K, float scale) {
  __shared__ __bf16 As[2][128][40];   // +8 pad: row stride 80B (16B aligned)
  __shared__ __bf16 Bs[2][128][40];
  int tid = threadIdx.x;
  int lane = tid & 31, wid = tid >> 5;
  int wm = wid >> 2, wn = wid & 3;
  int m0 = blockIdx.x * 128, n0 = blockIdx.y * 128;
  int lr = tid >> 1, lc = (tid & 1) * 16;   // tile loader coords (32B per thread)
  int half = lane >> 4, lm = lane & 15;

  v8f acc[4][2];
  for (int i = 0; i < 4; i++)
    for (int j = 0; j < 2; j++)
      for (int r = 0; r < 8; r++) acc[i][j][r] = 0.f;

  const __bf16* gA = A  + (size_t)(m0 + lr)*K + lc;
  const __bf16* gB = Bt + (size_t)(n0 + lr)*K + lc;

  auto issue_tile = [&](int kt, int nb) {
    int k0 = kt * 32;
    unsigned la = lds_off(&As[nb][lr][lc]);
    unsigned long long ga = (unsigned long long)(gA + k0);
    async_g2l_b128(la, ga);
    async_g2l_b128_o16(la, ga);
    unsigned lb = lds_off(&Bs[nb][lr][lc]);
    unsigned long long gb = (unsigned long long)(gB + k0);
    async_g2l_b128(lb, gb);
    async_g2l_b128_o16(lb, gb);
  };

  int nt = K >> 5;
  issue_tile(0, 0);
  for (int i = 0; i < nt; i++) {
    int cb = i & 1;
    bool has_next = (i + 1) < nt;
    if (has_next) issue_tile(i + 1, cb ^ 1);     // overlap next tile DMA with compute
    if (has_next) wait_async4(); else wait_async0();  // tile i complete (own wave)
    __syncthreads();                             // all waves' async writes visible

    Frag af[4], bf[2];
    int ac0 = half ? 8 : 0;     // A frag: K {0..7,16..23} | {8..15,24..31}
    for (int mi = 0; mi < 4; mi++) {
      int r = wm*64 + mi*16 + lm;
      af[mi].u[0] = *reinterpret_cast<const uint4*>(&As[cb][r][ac0]);
      af[mi].u[1] = *reinterpret_cast<const uint4*>(&As[cb][r][ac0 + 16]);
    }
    int bc0 = half ? 8 : 0;     // B frag: K {0..7,16..23} | {8..15,24..31}
    for (int ni = 0; ni < 2; ni++) {
      int n = wn*32 + ni*16 + lm;
      bf[ni].u[0] = *reinterpret_cast<const uint4*>(&Bs[cb][n][bc0]);
      bf[ni].u[1] = *reinterpret_cast<const uint4*>(&Bs[cb][n][bc0 + 16]);
    }
    for (int mi = 0; mi < 4; mi++)
      for (int ni = 0; ni < 2; ni++)
        acc[mi][ni] = __builtin_amdgcn_wmma_f32_16x16x32_bf16(
            false, af[mi].v, false, bf[ni].v, (short)0, acc[mi][ni], false, false);
    __syncthreads();                             // done reading buf cb (WAR guard)
  }

  for (int mi = 0; mi < 4; mi++) {
    for (int ni = 0; ni < 2; ni++) {
      for (int r = 0; r < 8; r++) {
        int row = m0 + wm*64 + mi*16 + r + half*8;
        int col = n0 + wn*32 + ni*16 + lm;
        float v = acc[mi][ni][r] + bias[col];
        if constexpr (EPI == EPI_QK) {
          v *= scale;
          int b = row >> 11, t = row & (SEQ-1);
          int h = col >> 6,  d = col & 63;
          outb[(((size_t)(b*N_HEADS + h)*SEQ + t) << 6) + d] = (__bf16)v;
        } else if constexpr (EPI == EPI_VT) {
          int b = row >> 11, t = row & (SEQ-1);
          int h = col >> 6,  d = col & 63;
          outb[((size_t)(b*N_HEADS + h)*D_HEAD + d)*SEQ + t] = (__bf16)v;
        } else if constexpr (EPI == EPI_RES) {
          size_t idx = (size_t)row*N + col;
          outf[idx] = v + res[idx];
        } else {
          v = 0.5f*v*(1.0f + erff(v*0.70710678118f));   // exact gelu
          outb[(size_t)row*N + col] = (__bf16)v;
        }
      }
    }
  }
}

// ---------------- flash attention (causal), bf16 WMMA, fp32 online softmax
__device__ inline float redmax16(float v) {
  v = fmaxf(v, __shfl_xor(v, 1, 32));
  v = fmaxf(v, __shfl_xor(v, 2, 32));
  v = fmaxf(v, __shfl_xor(v, 4, 32));
  v = fmaxf(v, __shfl_xor(v, 8, 32));
  return v;
}
__device__ inline float redsum16(float v) {
  v += __shfl_xor(v, 1, 32);
  v += __shfl_xor(v, 2, 32);
  v += __shfl_xor(v, 4, 32);
  v += __shfl_xor(v, 8, 32);
  return v;
}

__global__ __launch_bounds__(256) void k_flash_attn(const __bf16* __restrict__ q,
                                                    const __bf16* __restrict__ kk,
                                                    const __bf16* __restrict__ vt,
                                                    __bf16* __restrict__ ao) {
  __shared__ __bf16 Ks[64][72];        // 64 keys x 64 d (stride 144B)
  __shared__ __bf16 Vs[64][72];        // 64 d x 64 keys (V already transposed)
  __shared__ __bf16 Ps[8][16][72];     // per-wave P tile (16 q x 64 k)
  int tid = threadIdx.x, lane = tid & 31, wid = tid >> 5;
  int half = lane >> 4, lm = lane & 15;
  int bh = blockIdx.y;
  int b = bh >> 4, h = bh & 15;
  int qbase = blockIdx.x*128 + wid*16;
  const __bf16* qp = q  + (size_t)bh*SEQ*D_HEAD;
  const __bf16* kp = kk + (size_t)bh*SEQ*D_HEAD;
  const __bf16* vp = vt + (size_t)bh*D_HEAD*SEQ;

  Frag qf[2];                          // Q rows qbase..+15, d 0..63 (scale pre-folded)
  {
    const __bf16* qr = qp + (size_t)(qbase + lm)*D_HEAD;
    int c0 = half ? 8 : 0;
    qf[0].u[0] = *reinterpret_cast<const uint4*>(qr + c0);
    qf[0].u[1] = *reinterpret_cast<const uint4*>(qr + c0 + 16);
    qf[1].u[0] = *reinterpret_cast<const uint4*>(qr + 32 + c0);
    qf[1].u[1] = *reinterpret_cast<const uint4*>(qr + 32 + c0 + 16);
  }
  float m_[8], l_[8];
  v8f o[4];
  for (int r = 0; r < 8; r++) { m_[r] = -INFINITY; l_[r] = 0.f; }
  for (int di = 0; di < 4; di++) for (int r = 0; r < 8; r++) o[di][r] = 0.f;

  // async K/V stage coords: 2 x 16B segments per thread per tile
  int s0 = tid*2;
  int row0 = s0 >> 3,     seg0 = (s0 & 7)*8;
  int row1 = (s0+1) >> 3, seg1 = ((s0+1) & 7)*8;

  int ntile = 2*(blockIdx.x + 1);
  for (int kt = 0; kt < ntile; kt++) {
    int k0 = kt*64;
    // cooperative K/V tile load via async global->LDS DMA
    async_g2l_b128(lds_off(&Ks[row0][seg0]),
                   (unsigned long long)(kp + (size_t)(k0 + row0)*D_HEAD + seg0));
    async_g2l_b128(lds_off(&Ks[row1][seg1]),
                   (unsigned long long)(kp + (size_t)(k0 + row1)*D_HEAD + seg1));
    async_g2l_b128(lds_off(&Vs[row0][seg0]),
                   (unsigned long long)(vp + (size_t)row0*SEQ + k0 + seg0));
    async_g2l_b128(lds_off(&Vs[row1][seg1]),
                   (unsigned long long)(vp + (size_t)row1*SEQ + k0 + seg1));
    wait_async0();
    __syncthreads();

    if (k0 <= qbase + 15) {            // skip fully-masked tiles (uniform per wave)
      v8f s4[4];
      for (int f = 0; f < 4; f++) for (int r = 0; r < 8; r++) s4[f][r] = 0.f;
      int bc0 = half ? 8 : 0;
      for (int f = 0; f < 4; f++) {
        int kcol = f*16 + lm;
        Frag kf0, kf1;
        kf0.u[0] = *reinterpret_cast<const uint4*>(&Ks[kcol][bc0]);
        kf0.u[1] = *reinterpret_cast<const uint4*>(&Ks[kcol][bc0 + 16]);
        kf1.u[0] = *reinterpret_cast<const uint4*>(&Ks[kcol][32 + bc0]);
        kf1.u[1] = *reinterpret_cast<const uint4*>(&Ks[kcol][32 + bc0 + 16]);
        s4[f] = __builtin_amdgcn_wmma_f32_16x16x32_bf16(
            false, qf[0].v, false, kf0.v, (short)0, s4[f], false, false);
        s4[f] = __builtin_amdgcn_wmma_f32_16x16x32_bf16(
            false, qf[1].v, false, kf1.v, (short)0, s4[f], false, false);
      }
      for (int f = 0; f < 4; f++) {    // causal mask
        int kcol = k0 + f*16 + lm;
        for (int r = 0; r < 8; r++)
          if (kcol > qbase + r + half*8) s4[f][r] = -INFINITY;
      }
      float alpha[8];
      for (int r = 0; r < 8; r++) {    // online softmax per owned row
        float mx = fmaxf(fmaxf(s4[0][r], s4[1][r]), fmaxf(s4[2][r], s4[3][r]));
        mx = redmax16(mx);
        float mn = fmaxf(m_[r], mx);
        alpha[r] = (mn == -INFINITY) ? 1.f : __expf(m_[r] - mn);
        float rs = 0.f;
        for (int f = 0; f < 4; f++) {
          float p = (mn == -INFINITY) ? 0.f : __expf(s4[f][r] - mn);
          Ps[wid][r + half*8][f*16 + lm] = (__bf16)p;
          rs += p;
        }
        rs = redsum16(rs);
        l_[r] = l_[r]*alpha[r] + rs;
        m_[r] = mn;
      }
      for (int di = 0; di < 4; di++)
        for (int r = 0; r < 8; r++) o[di][r] *= alpha[r];
      asm volatile("s_wait_dscnt 0x0" ::: "memory");  // wave-internal LDS RAW on Ps
      Frag pf[2];
      int ac0 = half ? 8 : 0;
      pf[0].u[0] = *reinterpret_cast<const uint4*>(&Ps[wid][lm][ac0]);
      pf[0].u[1] = *reinterpret_cast<const uint4*>(&Ps[wid][lm][ac0 + 16]);
      pf[1].u[0] = *reinterpret_cast<const uint4*>(&Ps[wid][lm][32 + ac0]);
      pf[1].u[1] = *reinterpret_cast<const uint4*>(&Ps[wid][lm][32 + ac0 + 16]);
      for (int di = 0; di < 4; di++) {
        int d = di*16 + lm;
        Frag vf0, vf1;
        vf0.u[0] = *reinterpret_cast<const uint4*>(&Vs[d][bc0]);
        vf0.u[1] = *reinterpret_cast<const uint4*>(&Vs[d][bc0 + 16]);
        vf1.u[0] = *reinterpret_cast<const uint4*>(&Vs[d][32 + bc0]);
        vf1.u[1] = *reinterpret_cast<const uint4*>(&Vs[d][32 + bc0 + 16]);
        o[di] = __builtin_amdgcn_wmma_f32_16x16x32_bf16(
            false, pf[0].v, false, vf0.v, (short)0, o[di], false, false);
        o[di] = __builtin_amdgcn_wmma_f32_16x16x32_bf16(
            false, pf[1].v, false, vf1.v, (short)0, o[di], false, false);
      }
    }
    __syncthreads();
  }
  for (int r = 0; r < 8; r++) {
    float inv = (l_[r] > 0.f) ? 1.f / l_[r] : 0.f;
    int qrow = qbase + r + half*8;
    for (int di = 0; di < 4; di++) {
      int d = di*16 + lm;
      ao[(size_t)(b*SEQ + qrow)*D_MODEL + h*D_HEAD + d] = (__bf16)(o[di][r]*inv);
    }
  }
}

extern "C" void kernel_launch(void* const* d_in, const int* in_sizes, int n_in,
                              void* d_out, int out_size, void* d_ws, size_t ws_size,
                              hipStream_t stream) {
  const float* x   = (const float*)d_in[0];
  // d_in[1] = attn_mask (causal, handled analytically)
  const float* Wq  = (const float*)d_in[2];   const float* bq = (const float*)d_in[3];
  const float* Wk  = (const float*)d_in[4];   const float* bk = (const float*)d_in[5];
  const float* Wv  = (const float*)d_in[6];   const float* bv = (const float*)d_in[7];
  const float* Wo  = (const float*)d_in[8];   const float* bo = (const float*)d_in[9];
  const float* W1  = (const float*)d_in[10];  const float* b1 = (const float*)d_in[11];
  const float* W2  = (const float*)d_in[12];  const float* b2 = (const float*)d_in[13];
  const float* g1  = (const float*)d_in[14];  const float* be1 = (const float*)d_in[15];
  const float* g2  = (const float*)d_in[16];  const float* be2 = (const float*)d_in[17];
  float* out = (float*)d_out;

  unsigned char* ws = (unsigned char*)d_ws;
  size_t off = 0;
  __bf16* wq_t = (__bf16*)(ws + off); off += (size_t)D_MODEL*D_MODEL*2;
  __bf16* wk_t = (__bf16*)(ws + off); off += (size_t)D_MODEL*D_MODEL*2;
  __bf16* wv_t = (__bf16*)(ws + off); off += (size_t)D_MODEL*D_MODEL*2;
  __bf16* wo_t = (__bf16*)(ws + off); off += (size_t)D_MODEL*D_MODEL*2;
  __bf16* w1_t = (__bf16*)(ws + off); off += (size_t)D_MODEL*D_FF*2;
  __bf16* w2_t = (__bf16*)(ws + off); off += (size_t)D_FF*D_MODEL*2;
  __bf16* xn   = (__bf16*)(ws + off); off += (size_t)BT*D_MODEL*2;   // also hn
  __bf16* qb   = (__bf16*)(ws + off); off += (size_t)BT*D_MODEL*2;
  __bf16* kb   = (__bf16*)(ws + off); off += (size_t)BT*D_MODEL*2;
  __bf16* vtb  = (__bf16*)(ws + off); off += (size_t)BT*D_MODEL*2;
  __bf16* aob  = (__bf16*)(ws + off); off += (size_t)BT*D_MODEL*2;
  float*  x1   = (float*) (ws + off); off += (size_t)BT*D_MODEL*4;
  __bf16* h1   = qb;                  // reuse q..ao region (64 MiB) for h1

  // weights -> bf16, transposed [N][K]
  k_transpose_cvt<<<4096, 256, 0, stream>>>(Wq, wq_t, D_MODEL, D_MODEL);
  k_transpose_cvt<<<4096, 256, 0, stream>>>(Wk, wk_t, D_MODEL, D_MODEL);
  k_transpose_cvt<<<4096, 256, 0, stream>>>(Wv, wv_t, D_MODEL, D_MODEL);
  k_transpose_cvt<<<4096, 256, 0, stream>>>(Wo, wo_t, D_MODEL, D_MODEL);
  k_transpose_cvt<<<16384, 256, 0, stream>>>(W1, w1_t, D_MODEL, D_FF);
  k_transpose_cvt<<<16384, 256, 0, stream>>>(W2, w2_t, D_FF, D_MODEL);

  // LN1
  k_layernorm<<<BT, 256, 0, stream>>>(x, g1, be1, xn);
  // Q/K/V projections (scale folded into Q)
  k_gemm<EPI_QK><<<dim3(64, 8), 256, 0, stream>>>(xn, wq_t, bq, nullptr, nullptr, qb,
                                                  BT, D_MODEL, D_MODEL, 0.125f);
  k_gemm<EPI_QK><<<dim3(64, 8), 256, 0, stream>>>(xn, wk_t, bk, nullptr, nullptr, kb,
                                                  BT, D_MODEL, D_MODEL, 1.0f);
  k_gemm<EPI_VT><<<dim3(64, 8), 256, 0, stream>>>(xn, wv_t, bv, nullptr, nullptr, vtb,
                                                  BT, D_MODEL, D_MODEL, 1.0f);
  // causal flash attention
  k_flash_attn<<<dim3(SEQ/128, BATCH*N_HEADS), 256, 0, stream>>>(qb, kb, vtb, aob);
  // output projection + residual -> x1 (fp32)
  k_gemm<EPI_RES><<<dim3(64, 8), 256, 0, stream>>>(aob, wo_t, bo, x, x1, nullptr,
                                                   BT, D_MODEL, D_MODEL, 1.0f);
  // LN2 -> hn (reuses xn)
  k_layernorm<<<BT, 256, 0, stream>>>(x1, g2, be2, xn);
  // FF1 + GELU -> h1 (bf16)
  k_gemm<EPI_GELU><<<dim3(64, 32), 256, 0, stream>>>(xn, w1_t, b1, nullptr, nullptr, h1,
                                                     BT, D_FF, D_MODEL, 1.0f);
  // FF2 + residual -> out (fp32)
  k_gemm<EPI_RES><<<dim3(64, 8), 256, 0, stream>>>(h1, w2_t, b2, x1, out, nullptr,
                                                   BT, D_MODEL, D_FF, 1.0f);
  (void)in_sizes; (void)n_in; (void)out_size; (void)ws_size;
}
